// GAT_70153995813498
// MI455X (gfx1250) — matrix operand order repaired and checked by reading, not compiled, not run on hardware
//
#include <hip/hip_runtime.h>
#include <hip/hip_bf16.h>
#include <math.h>

typedef __attribute__((ext_vector_type(16))) __bf16 v16bf;
typedef __attribute__((ext_vector_type(8)))  float  v8f;
typedef __attribute__((ext_vector_type(4)))  float  v4f;

#define HEADS     4
#define DHEAD     64
#define FEAT_LD   256
#define NEG_SLOPE 0.2f
#define BN_EPS    1e-5f

// ---------------- weight repack: fp32 row-major [64 x C] -> bf16 WMMA B-fragments
// layout: [kt][ct][lane][j], j=0..15 per lane; K=kt*32+8*(lane>>4)+(j&7)+16*(j>>3)
__global__ void k_pack_b(const float* __restrict__ W, __bf16* __restrict__ out,
                         int ldw, int ctiles, int cvalid) {
  int idx = blockIdx.x * blockDim.x + threadIdx.x;
  int total = 2 * ctiles * 512;
  if (idx >= total) return;
  int j    = idx & 15;
  int lane = (idx >> 4) & 31;
  int tile = idx >> 9;
  int ct   = tile % ctiles;
  int kt   = tile / ctiles;
  int k = kt * 32 + 8 * (lane >> 4) + (j & 7) + 16 * (j >> 3);
  int c = ct * 16 + (lane & 15);
  float v = (c < cvalid) ? W[k * ldw + c] : 0.0f;
  out[idx] = (__bf16)v;
}

// ---------------- WMMA GEMM: C[M x ctiles*16] = A[M x 64] @ Bp, optional bias+relu
__global__ void k_gemm(const float* __restrict__ A, const __bf16* __restrict__ Bp,
                       const float* __restrict__ bias, float* __restrict__ C,
                       int M, int ctiles, int ldc, int cvalid, int relu) {
  int wave = (int)(blockIdx.x * blockDim.x + threadIdx.x) >> 5;
  int lane = threadIdx.x & 31;
  if (wave * 16 >= M) return;
  int rt   = wave;
  int row  = rt * 16 + (lane & 15);
  int half = lane >> 4;

  // build two A fragments (kt = 0,1) from fp32, converting to bf16 per ISA A-layout
  v16bf afrag[2];
#pragma unroll
  for (int kt = 0; kt < 2; ++kt) {
    const float* ap = A + (size_t)row * 64 + kt * 32 + 8 * half;
    v4f f0 = *(const v4f*)(ap + 0);
    v4f f1 = *(const v4f*)(ap + 4);
    v4f f2 = *(const v4f*)(ap + 16);
    v4f f3 = *(const v4f*)(ap + 20);
    v16bf a;
    a[0]  = (__bf16)f0[0]; a[1]  = (__bf16)f0[1]; a[2]  = (__bf16)f0[2]; a[3]  = (__bf16)f0[3];
    a[4]  = (__bf16)f1[0]; a[5]  = (__bf16)f1[1]; a[6]  = (__bf16)f1[2]; a[7]  = (__bf16)f1[3];
    a[8]  = (__bf16)f2[0]; a[9]  = (__bf16)f2[1]; a[10] = (__bf16)f2[2]; a[11] = (__bf16)f2[3];
    a[12] = (__bf16)f3[0]; a[13] = (__bf16)f3[1]; a[14] = (__bf16)f3[2]; a[15] = (__bf16)f3[3];
    afrag[kt] = a;
  }

  const v16bf* bp = (const v16bf*)Bp;
  for (int ct = 0; ct < ctiles; ++ct) {
    v8f acc = {};
#pragma unroll
    for (int kt = 0; kt < 2; ++kt) {
      v16bf b = bp[(kt * ctiles + ct) * 32 + lane];
      acc = __builtin_amdgcn_wmma_f32_16x16x32_bf16(
          false, afrag[kt], false, b, (short)0, acc, false, false);
    }
    int col = ct * 16 + (lane & 15);
    float bv = (bias != nullptr && col < cvalid) ? bias[col] : 0.0f;
#pragma unroll
    for (int r = 0; r < 8; ++r) {
      float v = acc[r] + bv;
      if (relu) v = fmaxf(v, 0.0f);
      C[(size_t)(rt * 16 + r + 8 * half) * ldc + col] = v;
    }
  }
}

// ---------------- attention logits el/er
__global__ void k_el_er(const float* __restrict__ feat, const float* __restrict__ al,
                        const float* __restrict__ ar, float* __restrict__ el,
                        float* __restrict__ er, int N) {
  int idx = blockIdx.x * blockDim.x + threadIdx.x;
  if (idx >= N * HEADS) return;
  int h = idx & 3, n = idx >> 2;
  const float* f  = feat + (size_t)n * FEAT_LD + h * DHEAD;
  const float* a1 = al + h * DHEAD;
  const float* a2 = ar + h * DHEAD;
  float sl = 0.f, sr = 0.f;
  for (int i = 0; i < DHEAD; i += 4) {
    v4f fv = *(const v4f*)(f + i);
    v4f lv = *(const v4f*)(a1 + i);
    v4f rv = *(const v4f*)(a2 + i);
    sl += fv[0]*lv[0] + fv[1]*lv[1] + fv[2]*lv[2] + fv[3]*lv[3];
    sr += fv[0]*rv[0] + fv[1]*rv[1] + fv[2]*rv[2] + fv[3]*rv[3];
  }
  el[idx] = sl; er[idx] = sr;
}

__global__ void k_init_md(float* __restrict__ m, float* __restrict__ denom, int n) {
  int i = blockIdx.x * blockDim.x + threadIdx.x;
  if (i < n) { m[i] = -INFINITY; denom[i] = 0.0f; }
}

__device__ inline void edge_sd(const int* __restrict__ ei, int E0, int e, int& s, int& d) {
  if (e < E0) { s = ei[e]; d = ei[E0 + e]; } else { s = e - E0; d = s; }
}

__device__ inline float leaky(float v) { return v > 0.f ? v : NEG_SLOPE * v; }

__device__ inline void atomicMaxF(float* addr, float val) {
  if (val >= 0.f) atomicMax((int*)addr, __float_as_int(val));
  else            atomicMin((unsigned int*)addr, (unsigned int)__float_as_int(val));
}

__global__ void k_edge_max(const int* __restrict__ ei, int E0, int Etot,
                           const float* __restrict__ el, const float* __restrict__ er,
                           float* __restrict__ m) {
  int e = blockIdx.x * blockDim.x + threadIdx.x;
  if (e >= Etot) return;
  int s, d; edge_sd(ei, E0, e, s, d);
  v4f lv = *(const v4f*)(el + (size_t)s * 4);
  v4f rv = *(const v4f*)(er + (size_t)d * 4);
#pragma unroll
  for (int h = 0; h < 4; ++h) atomicMaxF(&m[(size_t)d * 4 + h], leaky(lv[h] + rv[h]));
}

__global__ void k_edge_exp(const int* __restrict__ ei, int E0, int Etot,
                           const float* __restrict__ el, const float* __restrict__ er,
                           const float* __restrict__ m, float* __restrict__ ex,
                           float* __restrict__ denom) {
  int e = blockIdx.x * blockDim.x + threadIdx.x;
  if (e >= Etot) return;
  int s, d; edge_sd(ei, E0, e, s, d);
  v4f lv = *(const v4f*)(el + (size_t)s * 4);
  v4f rv = *(const v4f*)(er + (size_t)d * 4);
  v4f mv = *(const v4f*)(m  + (size_t)d * 4);
#pragma unroll
  for (int h = 0; h < 4; ++h) {
    float v = __expf(leaky(lv[h] + rv[h]) - mv[h]);
    ex[(size_t)e * 4 + h] = v;
    unsafeAtomicAdd(&denom[(size_t)d * 4 + h], v);
  }
}

__global__ void k_edge_aggr(const int* __restrict__ ei, int E0, int Etot,
                            const float* __restrict__ feat, const float* __restrict__ ex,
                            const float* __restrict__ denom, float* __restrict__ rst) {
  int idx = blockIdx.x * blockDim.x + threadIdx.x;
  if (idx >= Etot * 4) return;
  int e = idx >> 2, h = idx & 3;
  int s, d; edge_sd(ei, E0, e, s, d);
  float a = ex[idx] / denom[(size_t)d * 4 + h];
  const v4f* fp = (const v4f*)(feat + (size_t)s * FEAT_LD + h * DHEAD);
  float* rp = rst + (size_t)d * FEAT_LD + h * DHEAD;
#pragma unroll 4
  for (int i = 0; i < 16; ++i) {
    v4f f = fp[i];
    unsafeAtomicAdd(rp + i * 4 + 0, f[0] * a);
    unsafeAtomicAdd(rp + i * 4 + 1, f[1] * a);
    unsafeAtomicAdd(rp + i * 4 + 2, f[2] * a);
    unsafeAtomicAdd(rp + i * 4 + 3, f[3] * a);
  }
}

__global__ void k_head_mean(const float* __restrict__ rst, const float* __restrict__ bias,
                            float* __restrict__ hout, int N) {
  int idx = blockIdx.x * blockDim.x + threadIdx.x;
  if (idx >= N * DHEAD) return;
  int n = idx >> 6, dd = idx & 63;
  float acc = 0.f;
#pragma unroll
  for (int h = 0; h < HEADS; ++h)
    acc += fmaxf(rst[(size_t)n * FEAT_LD + h * DHEAD + dd] + bias[h * DHEAD + dd], 0.f);
  hout[idx] = acc * 0.25f;
}

// ---------------- BN stats over z [N x 208] (200 valid columns)
__global__ void k_colstats(const float* __restrict__ z, float* __restrict__ colsum,
                           float* __restrict__ colsumsq, int N, int rowsPer) {
  int c = threadIdx.x;
  if (c >= 208) return;
  int r0 = blockIdx.x * rowsPer;
  int r1 = min(N, r0 + rowsPer);
  float s = 0.f, q = 0.f;
  for (int r = r0; r < r1; ++r) {
    float v = z[(size_t)r * 208 + c];
    s += v; q += v * v;
  }
  unsafeAtomicAdd(&colsum[c], s);
  unsafeAtomicAdd(&colsumsq[c], q);
}

__global__ void k_bn_prep(const float* __restrict__ colsum, const float* __restrict__ colsumsq,
                          const float* __restrict__ g, const float* __restrict__ b,
                          const float* __restrict__ mw2, const float* __restrict__ mb2,
                          float* __restrict__ w2s, float* __restrict__ tshift, int N) {
  __shared__ float tsh[200];
  int c = threadIdx.x;
  float invN = 1.0f / (float)N;
  if (c < 200) {
    float mu  = colsum[c] * invN;
    float var = colsumsq[c] * invN - mu * mu;
    float s   = g[c] * rsqrtf(var + BN_EPS);
    float t   = b[c] - mu * s;
    tsh[c] = t;
    w2s[c * 2 + 0] = s * mw2[c * 2 + 0];
    w2s[c * 2 + 1] = s * mw2[c * 2 + 1];
  }
  __syncthreads();
  if (c < 2) {
    float acc = mb2[c];
    for (int i = 0; i < 200; ++i) acc += tsh[i] * mw2[i * 2 + c];
    tshift[c] = acc;
  }
}

__global__ void k_final(const float* __restrict__ z, const float* __restrict__ w2s,
                        const float* __restrict__ tshift, float* __restrict__ out, int N) {
  int n = blockIdx.x * blockDim.x + threadIdx.x;
  if (n >= N) return;
  float a0 = tshift[0], a1 = tshift[1];
  const float* zr = z + (size_t)n * 208;
  for (int c = 0; c < 200; c += 4) {
    v4f v = *(const v4f*)(zr + c);
#pragma unroll
    for (int i = 0; i < 4; ++i) {
      a0 += v[i] * w2s[(c + i) * 2 + 0];
      a1 += v[i] * w2s[(c + i) * 2 + 1];
    }
  }
  out[(size_t)n * 2 + 0] = a0;
  out[(size_t)n * 2 + 1] = a1;
}

extern "C" void kernel_launch(void* const* d_in, const int* in_sizes, int n_in,
                              void* d_out, int out_size, void* d_ws, size_t ws_size,
                              hipStream_t stream) {
  const float* features = (const float*)d_in[0];
  const int*   ei       = (const int*)d_in[1];
  const float* W1  = (const float*)d_in[2];
  const float* al1 = (const float*)d_in[3];
  const float* ar1 = (const float*)d_in[4];
  const float* b1  = (const float*)d_in[5];
  const float* W2  = (const float*)d_in[6];
  const float* al2 = (const float*)d_in[7];
  const float* ar2 = (const float*)d_in[8];
  const float* b2  = (const float*)d_in[9];
  const float* mw1 = (const float*)d_in[10];
  const float* mb1 = (const float*)d_in[11];
  const float* bng = (const float*)d_in[12];
  const float* bnb = (const float*)d_in[13];
  const float* mw2 = (const float*)d_in[14];
  const float* mb2 = (const float*)d_in[15];

  const int N    = in_sizes[0] / 64;
  const int E0   = in_sizes[1] / 2;
  const int Etot = E0 + N;

  char* ws = (char*)d_ws;
  size_t off = 0;
  auto alloc = [&](size_t bytes) -> void* {
    void* p = ws + off;
    off = (off + bytes + 255) & ~(size_t)255;
    return p;
  };

  float*  feat   = (float*)alloc((size_t)N * FEAT_LD * 4);   // also aliased as z
  float*  rst    = (float*)alloc((size_t)N * FEAT_LD * 4);
  float*  el     = (float*)alloc((size_t)N * 4 * 4);
  float*  er     = (float*)alloc((size_t)N * 4 * 4);
  float*  mbuf   = (float*)alloc((size_t)N * 4 * 4);
  float*  denom  = (float*)alloc((size_t)N * 4 * 4);
  float*  exbuf  = (float*)alloc((size_t)Etot * 4 * 4);
  float*  h1     = (float*)alloc((size_t)N * 64 * 4);
  float*  h2     = (float*)alloc((size_t)N * 64 * 4);
  __bf16* W1p    = (__bf16*)alloc(2 * 16 * 512 * sizeof(__bf16));
  __bf16* W2p    = (__bf16*)alloc(2 * 16 * 512 * sizeof(__bf16));
  __bf16* MW1p   = (__bf16*)alloc(2 * 13 * 512 * sizeof(__bf16));
  float*  colsum = (float*)alloc(208 * 4);
  float*  colsq  = (float*)alloc(208 * 4);
  float*  w2s    = (float*)alloc(208 * 2 * 4);
  float*  tshift = (float*)alloc(2 * 4);
  float*  z      = feat;  // alias: feat dead after layer 2 aggregation

  const int TPB = 256;
  // weight repack (tiny)
  k_pack_b<<<(2 * 16 * 512 + TPB - 1) / TPB, TPB, 0, stream>>>(W1, W1p, 256, 16, 256);
  k_pack_b<<<(2 * 16 * 512 + TPB - 1) / TPB, TPB, 0, stream>>>(W2, W2p, 256, 16, 256);
  k_pack_b<<<(2 * 13 * 512 + TPB - 1) / TPB, TPB, 0, stream>>>(mw1, MW1p, 200, 13, 200);

  const int waves     = (N + 15) / 16;           // 3125
  const int gemmBlks  = (waves + 7) / 8;          // 8 waves/block
  const int edgeBlks  = (Etot + TPB - 1) / TPB;
  const int aggrBlks  = (Etot * 4 + TPB - 1) / TPB;

  auto run_layer = [&](const float* hin, const __bf16* Wp, const float* al,
                       const float* ar, const float* bias, float* hout) {
    k_gemm<<<gemmBlks, TPB, 0, stream>>>(hin, Wp, nullptr, feat, N, 16, FEAT_LD, 256, 0);
    k_el_er<<<(N * 4 + TPB - 1) / TPB, TPB, 0, stream>>>(feat, al, ar, el, er, N);
    k_init_md<<<(N * 4 + TPB - 1) / TPB, TPB, 0, stream>>>(mbuf, denom, N * 4);
    hipMemsetAsync(rst, 0, (size_t)N * FEAT_LD * 4, stream);
    k_edge_max<<<edgeBlks, TPB, 0, stream>>>(ei, E0, Etot, el, er, mbuf);
    k_edge_exp<<<edgeBlks, TPB, 0, stream>>>(ei, E0, Etot, el, er, mbuf, exbuf, denom);
    k_edge_aggr<<<aggrBlks, TPB, 0, stream>>>(ei, E0, Etot, feat, exbuf, denom, rst);
    k_head_mean<<<(N * 64 + TPB - 1) / TPB, TPB, 0, stream>>>(rst, bias, hout, N);
  };

  run_layer(features, W1p, al1, ar1, b1, h1);
  run_layer(h1,       W2p, al2, ar2, b2, h2);

  // MLP head
  k_gemm<<<gemmBlks, TPB, 0, stream>>>(h2, MW1p, mb1, z, N, 13, 208, 200, 1);
  hipMemsetAsync(colsum, 0, 208 * 4, stream);
  hipMemsetAsync(colsq,  0, 208 * 4, stream);
  const int chunks = 128;
  k_colstats<<<chunks, TPB, 0, stream>>>(z, colsum, colsq, N, (N + chunks - 1) / chunks);
  k_bn_prep<<<1, TPB, 0, stream>>>(colsum, colsq, bng, bnb, mw2, mb2, w2s, tshift, N);
  k_final<<<(N + TPB - 1) / TPB, TPB, 0, stream>>>(z, w2s, tshift, (float*)d_out, N);
}
